// AttentionPool2d_17197049053383
// MI455X (gfx1250) — matrix operand order repaired and checked by reading, not compiled, not run on hardware
//
#include <hip/hip_runtime.h>
#include <hip/hip_bf16.h>

typedef __attribute__((ext_vector_type(16))) __bf16 v16bf;
typedef __attribute__((ext_vector_type(8)))  __bf16 v8bf;
typedef __attribute__((ext_vector_type(8)))  float  v8f;

#define B_   128
#define C_   1024
#define T_   257
#define TP   288            // T padded to multiple of 32 (zero rows 257..287)
#define H_   16
#define XROW 1024           // xfT row stride (elements)
#define XBAT (TP * XROW)    // per-batch stride of xfT

// ---------------- WMMA fragment helpers (wave32, contiguous K = half*16+0..15 in A and B) ----
// A: 16x32 bf16, lane L holds row (L&15); K pairs: half=L>>4 ->
//    v[0..7]  = K (half*16   .. half*16+7)
//    v[8..15] = K (half*16+8 .. half*16+15)
static __device__ __forceinline__ v16bf load_a_row(const __bf16* base, int ld, int lane) {
  int r = lane & 15, half = lane >> 4;
  const __bf16* p = base + r * ld + half * 16;
  v8bf c0 = *(const v8bf*)(p);
  v8bf c1 = *(const v8bf*)(p + 8);
  v16bf a;
#pragma unroll
  for (int i = 0; i < 8; ++i) { a[i] = c0[i]; a[8 + i] = c1[i]; }
  return a;
}
// A stored K-major (i.e. we need A[m][k] = base[k*ld + m])
static __device__ __forceinline__ v16bf load_a_col(const __bf16* base, int ld, int lane) {
  int r = lane & 15, half = lane >> 4;
  v16bf a;
#pragma unroll
  for (int i = 0; i < 8; ++i) {
    a[i]     = base[(half * 16 + i) * ld + r];
    a[8 + i] = base[(half * 16 + 8 + i) * ld + r];
  }
  return a;
}
// B: 32x16 bf16, lane holds column (L&15); K = half*16 + 0..15 contiguous.
// Storage where each column's K values are contiguous (col stride given).
static __device__ __forceinline__ v16bf load_b_col(const __bf16* base, long colstride, int lane) {
  int n = lane & 15, half = lane >> 4;
  return *(const v16bf*)(base + (long)n * colstride + half * 16);
}
// B stored row-major B[k][n] = base[k*ld + n]
static __device__ __forceinline__ v16bf load_b_row(const __bf16* base, int ld, int lane) {
  int n = lane & 15, half = lane >> 4;
  v16bf b;
#pragma unroll
  for (int i = 0; i < 16; ++i) b[i] = base[(half * 16 + i) * ld + n];
  return b;
}
static __device__ __forceinline__ v8f wmma_bf16(v16bf a, v16bf b, v8f c) {
  return __builtin_amdgcn_wmma_f32_16x16x32_bf16(false, a, false, b, (short)0, c, false, false);
}

// ---------------- K0: convert weights to bf16 --------------------------------
__global__ void k_wconv(const float* __restrict__ qkv_w, const float* __restrict__ c_w,
                        __bf16* __restrict__ qw, __bf16* __restrict__ cw) {
  int i = blockIdx.x * blockDim.x + threadIdx.x;
  int stride = gridDim.x * blockDim.x;
  const int N1 = 3072 * 1024, N2 = 1024 * 1024;
  for (int idx = i; idx < N1; idx += stride) qw[idx] = (__bf16)qkv_w[idx];
  for (int idx = i; idx < N2; idx += stride) cw[idx] = (__bf16)c_w[idx];
}

// ---------------- K1: build xfT[b][t][c] = x(+mean token) + pos_emb, bf16 ----
__global__ void k_prep(const float* __restrict__ x, const float* __restrict__ pos,
                       __bf16* __restrict__ xfT) {
  __shared__ float xs[32][257];   // stride 257 breaks bank conflicts
  int b = blockIdx.y, cb = blockIdx.x * 32;
  int tid = threadIdx.x;
  {                               // coalesced load of 32 channels x 256 spatial
    int c = tid >> 3;
    int s0 = (tid & 7) * 32;
    const float* src = x + ((long)(b * C_ + cb + c) * 256 + s0);
#pragma unroll
    for (int i = 0; i < 32; ++i) xs[c][s0 + i] = src[i];
  }
  __syncthreads();
  __bf16* xb = xfT + (long)b * XBAT;
  if (tid < 32) {                 // mean token -> row t=0
    int c = tid;
    float s = 0.f;
    for (int t = 0; t < 256; ++t) s += xs[c][t];
    float m = s * (1.0f / 256.0f) + pos[(cb + c) * T_ + 0];
    xb[cb + c] = (__bf16)m;
  }
  {                               // rows t=1..256 (transpose via LDS)
    int t = tid;                  // 0..255
    __bf16* dst = xb + (long)(t + 1) * XROW + cb;
    for (int c = 0; c < 32; ++c) {
      float v = xs[c][t] + pos[(cb + c) * T_ + (t + 1)];
      dst[c] = (__bf16)v;
    }
  }
  for (int idx = tid; idx < 31 * 32; idx += 256) {   // zero pad rows 257..287
    int t = 257 + (idx >> 5), c = idx & 31;
    xb[(long)t * XROW + cb + c] = (__bf16)0.0f;
  }
}

// ---------------- K2: q0[b][o] = Wq xf0 + bq  (M=1024,N=128,K=1024) ----------
__global__ void k_q0(const __bf16* __restrict__ qw, const __bf16* __restrict__ xfT,
                     const float* __restrict__ qkv_b, __bf16* __restrict__ q0) {
  int lane = threadIdx.x & 31;
  int wid = blockIdx.x * (blockDim.x >> 5) + (threadIdx.x >> 5);
  int ot = wid >> 3, bt = wid & 7;
  v8f acc = {};
  const __bf16* Abase = qw + (long)ot * 16 * 1024;          // Wq rows
  const __bf16* Bbase = xfT + (long)bt * 16 * XBAT;         // t=0 rows per batch
  for (int kk = 0; kk < 1024; kk += 32) {
    v16bf a = load_a_row(Abase + kk, 1024, lane);
    v16bf b = load_b_col(Bbase + kk, XBAT, lane);
    acc = wmma_bf16(a, b, acc);
  }
  int n = lane & 15, half = lane >> 4;
  __bf16* dst = q0 + (long)(bt * 16 + n) * 1024 + ot * 16 + half * 8;
  const float* bias = qkv_b + ot * 16 + half * 8;
#pragma unroll
  for (int j = 0; j < 8; ++j) dst[j] = (__bf16)(acc[j] + bias[j]);
}

// ---------------- K3: u[b][h][C] = Wk_h^T q0_h  (per h: M=1024,N=128,K=64) ---
__global__ void k_u(const __bf16* __restrict__ qw, const __bf16* __restrict__ q0,
                    __bf16* __restrict__ u) {
  int lane = threadIdx.x & 31;
  int wid = blockIdx.x * (blockDim.x >> 5) + (threadIdx.x >> 5);
  int h = wid >> 9;
  int Ct = (wid >> 3) & 63;
  int bt = wid & 7;
  v8f acc = {};
  const __bf16* Abase = qw + (long)(1024 + h * 64) * 1024 + Ct * 16;  // K-major
  const __bf16* Bbase = q0 + (long)bt * 16 * 1024 + h * 64;
#pragma unroll
  for (int kk = 0; kk < 64; kk += 32) {
    v16bf a = load_a_col(Abase + (long)kk * 1024, 1024, lane);
    v16bf b = load_b_col(Bbase + kk, 1024, lane);
    acc = wmma_bf16(a, b, acc);
  }
  int n = lane & 15, half = lane >> 4;
  __bf16* dst = u + (long)(bt * 16 + n) * (H_ * 1024) + h * 1024 + Ct * 16 + half * 8;
#pragma unroll
  for (int j = 0; j < 8; ++j) dst[j] = (__bf16)acc[j];
}

// ---------------- K4: per batch: scores -> softmax -> y ----------------------
#define SST 280
#define WST 296
__global__ void k_attn(const __bf16* __restrict__ xfT, const __bf16* __restrict__ u,
                       __bf16* __restrict__ y) {
  __shared__ float  sc[16][SST];      // scores [h][t]
  __shared__ __bf16 wb[16][WST];      // softmax weights [h][t], zero-padded
  __shared__ __bf16 xch[32][512];     // 32 t-rows x 512 C-cols staging
  int b = blockIdx.x;
  int lane = threadIdx.x & 31;
  int w = threadIdx.x >> 5;           // wave 0..15
  const __bf16* xb = xfT + (long)b * XBAT;
  const __bf16* ub = u + (long)b * (H_ * 1024);

  // phase 1: scores^T tiles: D[m=t][n=h] = xfT x u^T (K=C=1024), 17 t-tiles
  for (int tile = w; tile < 17; tile += 16) {
    v8f acc = {};
    const __bf16* Abase = xb + (long)tile * 16 * XROW;
    for (int kk = 0; kk < 1024; kk += 32) {
      v16bf a  = load_a_row(Abase + kk, XROW, lane);
      v16bf bb = load_b_col(ub + kk, 1024, lane);
      acc = wmma_bf16(a, bb, acc);
    }
    int n = lane & 15, half = lane >> 4;
    float* dst = &sc[n][tile * 16 + half * 8];
#pragma unroll
    for (int j = 0; j < 8; ++j) dst[j] = acc[j];
  }
  __syncthreads();

  // phase 2: softmax over t (scale 1/sqrt(64)=0.125); wave w -> head w
  {
    int h = w;
    float mx = -3.4e38f;
    for (int t = lane; t < T_; t += 32) mx = fmaxf(mx, sc[h][t] * 0.125f);
#pragma unroll
    for (int off = 16; off > 0; off >>= 1) mx = fmaxf(mx, __shfl_xor(mx, off, 32));
    float sum = 0.f;
    for (int t = lane; t < T_; t += 32) {
      float e = __expf(sc[h][t] * 0.125f - mx);
      sc[h][t] = e;
      sum += e;
    }
#pragma unroll
    for (int off = 16; off > 0; off >>= 1) sum += __shfl_xor(sum, off, 32);
    float inv = 1.0f / sum;
    for (int t = lane; t < TP; t += 32)
      wb[h][t] = (t < T_) ? (__bf16)(sc[h][t] * inv) : (__bf16)0.0f;
  }
  __syncthreads();

  // phase 3: y[h][C] = w x xfT  (M=16 heads, N=1024 C, K=288 t)
  // B-operand chunks staged into LDS with CDNA5 async global->LDS DMA
  // (bypasses VGPRs, tracked by ASYNCcnt, overlaps with WMMA issue).
  __bf16* yb = y + (long)b * (H_ * 1024);
  for (int ch = 0; ch < 2; ++ch) {
    v8f acc0 = {}, acc1 = {};
    for (int kc = 0; kc < 9; ++kc) {
      __syncthreads();   // previous chunk fully consumed before overwrite
      {  // async copy: t rows kc*32.., C cols ch*512.. (64 B per thread)
        int row = threadIdx.x >> 4;
        int c0  = (threadIdx.x & 15) * 32;
        const __bf16* gsrc = xb + (long)(kc * 32 + row) * XROW + ch * 512 + c0;
        unsigned long long gaddr = (unsigned long long)(uintptr_t)gsrc;
        // low 32 bits of a generic shared pointer == LDS byte offset (ISA 10.2)
        unsigned lds_addr = (unsigned)(uintptr_t)(void*)&xch[row][c0];
        asm volatile(
            "global_load_async_to_lds_b128 %0, %1, off\n\t"
            "global_load_async_to_lds_b128 %0, %1, off offset:16\n\t"
            "global_load_async_to_lds_b128 %0, %1, off offset:32\n\t"
            "global_load_async_to_lds_b128 %0, %1, off offset:48"
            :: "v"(lds_addr), "v"(gaddr) : "memory");
      }
      asm volatile("s_wait_asynccnt 0x0" ::: "memory");  // own DMAs done
      __syncthreads();                                   // all waves' DMAs visible
      v16bf a  = load_a_row(&wb[0][0] + kc * 32, WST, lane);
      v16bf b0 = load_b_row(&xch[0][0] + (w * 2) * 16, 512, lane);
      v16bf b1 = load_b_row(&xch[0][0] + (w * 2 + 1) * 16, 512, lane);
      acc0 = wmma_bf16(a, b0, acc0);
      acc1 = wmma_bf16(a, b1, acc1);
    }
    int n = lane & 15, half = lane >> 4;
    int Cg0 = ch * 512 + (w * 2) * 16 + n;
#pragma unroll
    for (int j = 0; j < 8; ++j) {
      int m = half * 8 + j;                 // head index
      yb[m * 1024 + Cg0]      = (__bf16)acc0[j];
      yb[m * 1024 + Cg0 + 16] = (__bf16)acc1[j];
    }
  }
}

// ---------------- K5: a0[b][c] = Wv y + bv  (per h: M=64,N=128,K=1024) ------
__global__ void k_a0(const __bf16* __restrict__ qw, const __bf16* __restrict__ y,
                     const float* __restrict__ qkv_b, __bf16* __restrict__ a0) {
  int lane = threadIdx.x & 31;
  int wid = blockIdx.x * (blockDim.x >> 5) + (threadIdx.x >> 5);
  int h = wid >> 5;
  int cct = (wid >> 3) & 3;
  int bt = wid & 7;
  v8f acc = {};
  const __bf16* Abase = qw + (long)(2048 + h * 64 + cct * 16) * 1024;
  const __bf16* Bbase = y + (long)bt * 16 * (H_ * 1024) + h * 1024;
  for (int kk = 0; kk < 1024; kk += 32) {
    v16bf a = load_a_row(Abase + kk, 1024, lane);
    v16bf b = load_b_col(Bbase + kk, H_ * 1024, lane);
    acc = wmma_bf16(a, b, acc);
  }
  int n = lane & 15, half = lane >> 4;
  int m0 = h * 64 + cct * 16 + half * 8;
  __bf16* dst = a0 + (long)(bt * 16 + n) * 1024 + m0;
  const float* bias = qkv_b + 2048 + m0;
#pragma unroll
  for (int j = 0; j < 8; ++j) dst[j] = (__bf16)(acc[j] + bias[j]);
}

// ---------------- K6: out[b][o] = c_w a0 + c_b  (M=1024,N=128,K=1024), fp32 -
__global__ void k_out(const __bf16* __restrict__ cw, const __bf16* __restrict__ a0,
                      const float* __restrict__ c_b, float* __restrict__ out) {
  int lane = threadIdx.x & 31;
  int wid = blockIdx.x * (blockDim.x >> 5) + (threadIdx.x >> 5);
  int ot = wid >> 3, bt = wid & 7;
  v8f acc = {};
  const __bf16* Abase = cw + (long)ot * 16 * 1024;
  const __bf16* Bbase = a0 + (long)bt * 16 * 1024;
  for (int kk = 0; kk < 1024; kk += 32) {
    v16bf a = load_a_row(Abase + kk, 1024, lane);
    v16bf b = load_b_col(Bbase + kk, 1024, lane);
    acc = wmma_bf16(a, b, acc);
  }
  int n = lane & 15, half = lane >> 4;
  float* dst = out + (long)(bt * 16 + n) * 1024 + ot * 16 + half * 8;
  const float* bias = c_b + ot * 16 + half * 8;
#pragma unroll
  for (int j = 0; j < 8; ++j) dst[j] = acc[j] + bias[j];
}

// ---------------- launch -----------------------------------------------------
extern "C" void kernel_launch(void* const* d_in, const int* in_sizes, int n_in,
                              void* d_out, int out_size, void* d_ws, size_t ws_size,
                              hipStream_t stream) {
  (void)in_sizes; (void)n_in; (void)out_size; (void)ws_size;
  const float* x     = (const float*)d_in[0];
  const float* pos   = (const float*)d_in[1];
  const float* qkv_w = (const float*)d_in[2];
  const float* qkv_b = (const float*)d_in[3];
  const float* c_w   = (const float*)d_in[4];
  const float* c_b   = (const float*)d_in[5];
  float* out = (float*)d_out;

  char* ws = (char*)d_ws;
  __bf16* qw  = (__bf16*)(ws + 0);          //  6.29 MB qkv_w bf16
  __bf16* cw  = (__bf16*)(ws + 6291456);    //  2.10 MB c_w bf16
  __bf16* xfT = (__bf16*)(ws + 8388608);    // 75.50 MB xfT [B][288][1024]
  __bf16* q0  = (__bf16*)(ws + 83886080);   //  0.26 MB
  __bf16* u   = (__bf16*)(ws + 84148224);   //  4.19 MB
  __bf16* y   = (__bf16*)(ws + 88342528);   //  4.19 MB
  __bf16* a0  = (__bf16*)(ws + 92536832);   //  0.26 MB  (total ~88.5 MB)

  k_wconv<<<dim3(1024), dim3(256), 0, stream>>>(qkv_w, c_w, qw, cw);
  k_prep <<<dim3(32, 128), dim3(256), 0, stream>>>(x, pos, xfT);
  k_q0   <<<dim3(64),  dim3(256), 0, stream>>>(qw, xfT, qkv_b, q0);
  k_u    <<<dim3(1024), dim3(256), 0, stream>>>(qw, q0, u);
  k_attn <<<dim3(128), dim3(512), 0, stream>>>(xfT, u, y);
  k_a0   <<<dim3(64),  dim3(256), 0, stream>>>(qw, y, qkv_b, a0);
  k_out  <<<dim3(64),  dim3(256), 0, stream>>>(cw, a0, c_b, out);
}
